// Block_34969623724753
// MI455X (gfx1250) — matrix operand order repaired and checked by reading, not compiled, not run on hardware
//
#include <hip/hip_runtime.h>
#include <hip/hip_bf16.h>

typedef __attribute__((ext_vector_type(16))) _Float16 v16h;
typedef __attribute__((ext_vector_type(8)))  _Float16 v8h;
typedef __attribute__((ext_vector_type(8)))  float    v8f;
typedef __attribute__((ext_vector_type(4)))  int      v4i;

typedef __attribute__((address_space(1))) v4i gv4i;   // global
typedef __attribute__((address_space(3))) v4i lv4i;   // LDS

union F16Frag { v16h v; v8h h2[2]; _Float16 e[16]; };

// ---------------------------------------------------------------------------
// Async global->LDS copy (CDNA5 GLOBAL_LOAD_ASYNC_TO_LDS_B128, ASYNCcnt),
// guarded so it degrades to a vector load+store if the builtin is absent.
// ---------------------------------------------------------------------------
#if defined(__has_builtin)
#if __has_builtin(__builtin_amdgcn_global_load_async_to_lds_b128)
#define HAS_ASYNC_LDS 1
#endif
#endif

__device__ inline void cp_b128(void* lds, const void* g) {
#ifdef HAS_ASYNC_LDS
    __builtin_amdgcn_global_load_async_to_lds_b128((gv4i*)g, (lv4i*)lds, 0, 0);
#else
    *reinterpret_cast<v8h*>(lds) = *reinterpret_cast<const v8h*>(g);
#endif
}

__device__ inline void async_wait() {
#ifdef HAS_ASYNC_LDS
#if __has_builtin(__builtin_amdgcn_s_wait_asynccnt)
    __builtin_amdgcn_s_wait_asynccnt(0);
#else
    asm volatile("s_wait_asynccnt 0x0" ::: "memory");
#endif
#endif
}

// ---------------------------------------------------------------------------
// Fragment loaders for 16-bit WMMA; A and B share one contiguous K order (K = 16h + j).
// A-matrix 16x32 (MxK): lane holds row M=lane&15; half j -> K = j + (lane>=16?16:0)
// ---------------------------------------------------------------------------
__device__ inline v16h load_a_frag(const _Float16* base, int strideHalves, int lane) {
    int m    = lane & 15;
    int koff = (lane >= 16) ? 16 : 0;
    const _Float16* p = base + (size_t)m * strideHalves + koff;
    F16Frag f;
    f.h2[0] = *reinterpret_cast<const v8h*>(p);        // K = koff .. koff+7
    f.h2[1] = *reinterpret_cast<const v8h*>(p + 8);    // K = koff+8 .. koff+15
    return f.v;
}

// B-matrix 32x16 (KxN) from an N-major (transposed) tile [n][k]:
// lane holds col N=lane&15; half j -> K = j + (lane>=16?16:0)  (16 contiguous halves)
__device__ inline v16h load_b_fragT(const _Float16* base, int strideHalves, int lane) {
    int n    = lane & 15;
    int koff = (lane >= 16) ? 16 : 0;
    return *reinterpret_cast<const v16h*>(base + (size_t)n * strideHalves + koff);
}

// B-matrix 32x16 from a K-major tile [k][n] (strided gather).
__device__ inline v16h load_b_frag_strided(const _Float16* base, int strideHalves, int lane) {
    int n    = lane & 15;
    int koff = (lane >= 16) ? 16 : 0;
    F16Frag f;
    #pragma unroll
    for (int j = 0; j < 16; ++j)
        f.e[j] = base[(size_t)(koff + j) * strideHalves + n];
    return f.v;
}

__device__ inline v8f wmma_f16(v16h a, v16h b, v8f c) {
    return __builtin_amdgcn_wmma_f32_16x16x32_f16(false, a, false, b, (short)0, c, false, false);
}

// ---------------------------------------------------------------------------
// Weight prep: fp32 (KxN) -> fp16 transposed (NxK).  One-time; amortized over
// all 73 M-tiles, and makes every GEMM LDS stage a 16-byte vector copy.
// ---------------------------------------------------------------------------
__global__ void convert_transpose_kernel(const float* __restrict__ in, _Float16* __restrict__ out,
                                         int K, int N) {
    int i = blockIdx.x * blockDim.x + threadIdx.x;
    if (i < K * N) {
        int n = i / K, k = i - n * K;
        out[(size_t)n * K + k] = (_Float16)in[(size_t)k * N + n];
    }
}

// ---------------------------------------------------------------------------
// LayerNorm over D=768, one block (256 threads) per token; f32 in -> f16 out
// ---------------------------------------------------------------------------
__global__ __launch_bounds__(256)
void layernorm_f16_kernel(const float* __restrict__ x, const float* __restrict__ g,
                          const float* __restrict__ b, _Float16* __restrict__ out) {
    const int t   = blockIdx.x;
    const int tid = threadIdx.x;
    const float* row = x + (size_t)t * 768;
    float v[3], s = 0.f, ss = 0.f;
    #pragma unroll
    for (int i = 0; i < 3; ++i) { v[i] = row[tid + i * 256]; s += v[i]; ss += v[i] * v[i]; }
    __shared__ float sbuf[256], ssbuf[256];
    sbuf[tid] = s; ssbuf[tid] = ss;
    __syncthreads();
    for (int off = 128; off > 0; off >>= 1) {
        if (tid < off) { sbuf[tid] += sbuf[tid + off]; ssbuf[tid] += ssbuf[tid + off]; }
        __syncthreads();
    }
    const float mu  = sbuf[0] * (1.0f / 768.0f);
    const float var = ssbuf[0] * (1.0f / 768.0f) - mu * mu;
    const float inv = rsqrtf(var + 1e-6f);
    #pragma unroll
    for (int i = 0; i < 3; ++i) {
        int c = tid + i * 256;
        out[(size_t)t * 768 + c] = (_Float16)((v[i] - mu) * inv * g[c] + b[c]);
    }
}

// ---------------------------------------------------------------------------
// Tiled WMMA GEMM: C = A(MxK,f16) @ W^T(NxK,f16 N-major) with epilogue.
// Block tile 128x64, 8 waves each own 32x32 (4 wmma / K-step of 32).
// All LDS staging is 16B async (or vector) copies; no transpose on stage.
// ---------------------------------------------------------------------------
__global__ __launch_bounds__(256)
void gemm_wmma_kernel(const _Float16* __restrict__ A, const _Float16* __restrict__ WT,
                      int M, int K, int Nout,
                      const float* __restrict__ bias, const float* __restrict__ resid,
                      float* __restrict__ outF, _Float16* __restrict__ outH, int act) {
    __shared__ alignas(32) _Float16 As[128 * 32];
    __shared__ alignas(32) _Float16 BsT[64 * 32];

    const int bm = blockIdx.x, bn = blockIdx.y;
    const int tid = threadIdx.x, lane = tid & 31, wave = tid >> 5;
    const int wm = (wave >> 1) * 32;   // 0,32,64,96
    const int wn = (wave & 1) * 32;    // 0,32

    v8f acc[2][2] = {};

    for (int k0 = 0; k0 < K; k0 += 32) {
        __syncthreads();
        // Stage A tile (128x32): two b128 copies per thread, zero-fill OOB rows.
        #pragma unroll
        for (int i = 0; i < 2; ++i) {
            int idx = i * 256 + tid;          // 0..511 (each = 8 halves)
            int m = idx >> 2, k8 = (idx & 3) * 8;
            int mg = bm * 128 + m;
            if (mg < M) cp_b128(&As[m * 32 + k8], A + (size_t)mg * K + k0 + k8);
            else        *reinterpret_cast<v8h*>(&As[m * 32 + k8]) = (v8h){};
        }
        // Stage B tile (64x32, N-major): one b128 copy per thread.
        {
            int n = tid >> 2, k8 = (tid & 3) * 8;
            cp_b128(&BsT[n * 32 + k8], WT + (size_t)(bn * 64 + n) * K + k0 + k8);
        }
        // Prefetch next K-step sources into cache.
        if (k0 + 32 < K) {
            int mgp = bm * 128 + (tid & 127);
            if (mgp < M) __builtin_prefetch(A + (size_t)mgp * K + k0 + 32, 0, 0);
            __builtin_prefetch(WT + (size_t)(bn * 64 + (tid & 63)) * K + k0 + 32, 0, 0);
        }
        async_wait();
        __syncthreads();

        v16h a0 = load_a_frag(&As[(wm +  0) * 32], 32, lane);
        v16h a1 = load_a_frag(&As[(wm + 16) * 32], 32, lane);
        v16h b0 = load_b_fragT(&BsT[(wn +  0) * 32], 32, lane);
        v16h b1 = load_b_fragT(&BsT[(wn + 16) * 32], 32, lane);
        acc[0][0] = wmma_f16(a0, b0, acc[0][0]);
        acc[0][1] = wmma_f16(a0, b1, acc[0][1]);
        acc[1][0] = wmma_f16(a1, b0, acc[1][0]);
        acc[1][1] = wmma_f16(a1, b1, acc[1][1]);
    }

    // Epilogue. C/D layout: VGPR r -> (M = r + (lane>=16?8:0), N = lane&15).
    const int moff = (lane >= 16) ? 8 : 0;
    #pragma unroll
    for (int mt = 0; mt < 2; ++mt)
    #pragma unroll
    for (int nt = 0; nt < 2; ++nt)
    #pragma unroll
    for (int r = 0; r < 8; ++r) {
        int m = bm * 128 + wm + mt * 16 + r + moff;
        int n = bn * 64 + wn + nt * 16 + (lane & 15);
        if (m < M) {
            float v = acc[mt][nt][r];
            if (bias)     v += bias[n];
            if (act == 1) v = 0.5f * v * (1.0f + erff(v * 0.70710678118f));  // exact GELU
            if (resid)    v += resid[(size_t)m * Nout + n];
            size_t o = (size_t)m * Nout + n;
            if (outF) outF[o] = v;
            if (outH) outH[o] = (_Float16)v;
        }
    }
}

// ---------------------------------------------------------------------------
// Fused attention with softmax over the HEADS axis (axis=1), exact streaming:
// per (b, 16-q tile), loop k in chunks of 32.  All 12 heads' dp for a (q,k)
// element are materialized before softmax, so no running rescale is needed.
// 128 threads = 4 waves, 3 heads each.  Zero-padded OOB k/v rows contribute 0.
// Q/K/V all stage row-major with b128 async copies; the attn@V B fragment is
// gathered strided from the row-major V tile.
// ---------------------------------------------------------------------------
#define ATTN_SMEM ((16*768 + 32*768 + 32*768) * 2 + 12*16*32 * 4 + 12*16*32 * 2)

__global__ __launch_bounds__(128)
void attn_fused_kernel(const _Float16* __restrict__ qkv, _Float16* __restrict__ wa, int Nn) {
    extern __shared__ char smem[];
    _Float16* q_lds  = (_Float16*)smem;               // 16 x 768
    _Float16* k_lds  = q_lds + 16 * 768;              // 32 x 768
    _Float16* v_lds  = k_lds + 32 * 768;              // 32 x 768 (row-major)
    float*    dp_lds = (float*)(v_lds + 32 * 768);    // 12 x 16 x 32
    _Float16* at_lds = (_Float16*)(dp_lds + 12*16*32);// 12 x 16 x 32

    const int qt = blockIdx.x, b = blockIdx.y;
    const int tid = threadIdx.x, lane = tid & 31, wave = tid >> 5;
    const float scale = 0.125f;  // DH^-0.5

    // Stage Q tile once (16x768 = 1536 b128 copies).
    #pragma unroll 4
    for (int i = 0; i < 12; ++i) {
        int idx = i * 128 + tid;
        int qr = idx / 96, c8 = (idx % 96) * 8;
        int qg = qt * 16 + qr;
        if (qg < Nn) cp_b128(&q_lds[qr * 768 + c8], qkv + ((size_t)(b * Nn + qg)) * 2304 + c8);
        else         *reinterpret_cast<v8h*>(&q_lds[qr * 768 + c8]) = (v8h){};
    }

    v8f wacc[3][4] = {};
    const int moff = (lane >= 16) ? 8 : 0;

    for (int kc = 0; kc < 19; ++kc) {            // 19 * 32 = 608 >= 577
        const int k0 = kc * 32;
        __syncthreads();
        // Stage K and V chunks (32x768 each = 3072 b128 copies apiece).
        #pragma unroll 4
        for (int i = 0; i < 24; ++i) {
            int idx = i * 128 + tid;
            int kr = idx / 96, c8 = (idx % 96) * 8;
            int kg = k0 + kr;
            if (kg < Nn) {
                size_t base = ((size_t)(b * Nn + kg)) * 2304;
                cp_b128(&k_lds[kr * 768 + c8], qkv + base + 768  + c8);
                cp_b128(&v_lds[kr * 768 + c8], qkv + base + 1536 + c8);
            } else {
                *reinterpret_cast<v8h*>(&k_lds[kr * 768 + c8]) = (v8h){};
                *reinterpret_cast<v8h*>(&v_lds[kr * 768 + c8]) = (v8h){};
            }
        }
        async_wait();
        __syncthreads();

        // dp = q . k^T  (per head: K=64 via two chained wmma, N=32 via two tiles)
        #pragma unroll
        for (int hh = 0; hh < 3; ++hh) {
            const int h = wave * 3 + hh;
            v8f dp[2] = {};
            #pragma unroll
            for (int c2 = 0; c2 < 64; c2 += 32) {
                v16h a = load_a_frag(&q_lds[h * 64 + c2], 768, lane);
                #pragma unroll
                for (int nt = 0; nt < 2; ++nt) {
                    int kcol = nt * 16 + (lane & 15);
                    v16h bf = *reinterpret_cast<const v16h*>(
                        &k_lds[(size_t)kcol * 768 + h * 64 + c2 + ((lane >= 16) ? 16 : 0)]);
                    dp[nt] = wmma_f16(a, bf, dp[nt]);
                }
            }
            #pragma unroll
            for (int nt = 0; nt < 2; ++nt)
            #pragma unroll
            for (int r = 0; r < 8; ++r) {
                int qq = r + moff;
                int kk = nt * 16 + (lane & 15);
                dp_lds[h * 512 + qq * 32 + kk] = dp[nt][r] * scale;
            }
        }
        __syncthreads();

        // Softmax across heads (axis of size 12), per (q,k) element.
        #pragma unroll
        for (int i = 0; i < 4; ++i) {
            int idx = i * 128 + tid;             // 0..511
            int qq = idx >> 5, kk = idx & 31;
            float mx = -1e30f;
            #pragma unroll
            for (int h = 0; h < 12; ++h) mx = fmaxf(mx, dp_lds[h * 512 + qq * 32 + kk]);
            float ev[12], s = 0.f;
            #pragma unroll
            for (int h = 0; h < 12; ++h) { ev[h] = __expf(dp_lds[h * 512 + qq * 32 + kk] - mx); s += ev[h]; }
            float inv = 1.0f / s;
            #pragma unroll
            for (int h = 0; h < 12; ++h) at_lds[h * 512 + qq * 32 + kk] = (_Float16)(ev[h] * inv);
        }
        __syncthreads();

        // wa += attn(16x32) @ v(32x64), full K=32 wmma per n-tile.
        #pragma unroll
        for (int hh = 0; hh < 3; ++hh) {
            const int h = wave * 3 + hh;
            v16h a = load_a_frag(&at_lds[h * 512], 32, lane);
            #pragma unroll
            for (int nt = 0; nt < 4; ++nt) {
                v16h bf = load_b_frag_strided(&v_lds[h * 64 + nt * 16], 768, lane);
                wacc[hh][nt] = wmma_f16(a, bf, wacc[hh][nt]);
            }
        }
    }

    // Store wa in (b, n, h*64+dh) order (== transpose(0,2,1,3).reshape).
    #pragma unroll
    for (int hh = 0; hh < 3; ++hh) {
        const int h = wave * 3 + hh;
        #pragma unroll
        for (int nt = 0; nt < 4; ++nt)
        #pragma unroll
        for (int r = 0; r < 8; ++r) {
            int qq = r + moff;
            int qg = qt * 16 + qq;
            if (qg < Nn)
                wa[((size_t)(b * Nn + qg)) * 768 + h * 64 + nt * 16 + (lane & 15)]
                    = (_Float16)wacc[hh][nt][r];
        }
    }
}

// ---------------------------------------------------------------------------
extern "C" void kernel_launch(void* const* d_in, const int* in_sizes, int n_in,
                              void* d_out, int out_size, void* d_ws, size_t ws_size,
                              hipStream_t stream) {
    (void)in_sizes; (void)n_in; (void)out_size; (void)ws_size;
    const int Bb = 16, Nn = 577, D = 768, HID = 3072;
    const int M = Bb * Nn;                 // 9232 tokens
    const int MT = (M + 127) / 128;        // 73

    const float* x      = (const float*)d_in[0];
    const float* ln1_g  = (const float*)d_in[1];
    const float* ln1_b  = (const float*)d_in[2];
    const float* qkv_w  = (const float*)d_in[3];
    const float* qkv_b  = (const float*)d_in[4];
    const float* proj_w = (const float*)d_in[5];
    const float* proj_b = (const float*)d_in[6];
    const float* ln2_g  = (const float*)d_in[7];
    const float* ln2_b  = (const float*)d_in[8];
    const float* fc1_w  = (const float*)d_in[9];
    const float* fc1_b  = (const float*)d_in[10];
    const float* fc2_w  = (const float*)d_in[11];
    const float* fc2_b  = (const float*)d_in[12];

    // Carve workspace (256B aligned chunks).
    char* w = (char*)d_ws;
    auto carve = [&](size_t bytes) { char* p = w; w += (bytes + 255) & ~(size_t)255; return p; };
    _Float16* qkvwT = (_Float16*)carve((size_t)D * 3 * D * 2);   // (3D x D) N-major
    _Float16* projwT= (_Float16*)carve((size_t)D * D * 2);       // (D x D)
    _Float16* fc1wT = (_Float16*)carve((size_t)D * HID * 2);     // (HID x D)
    _Float16* fc2wT = (_Float16*)carve((size_t)HID * D * 2);     // (D x HID)
    _Float16* h1    = (_Float16*)carve((size_t)M * D * 2);
    _Float16* qkv16 = (_Float16*)carve((size_t)M * 3 * D * 2);
    _Float16* wa16  = (_Float16*)carve((size_t)M * D * 2);
    float*    xmid  = (float*)   carve((size_t)M * D * 4);
    _Float16* h2    = (_Float16*)carve((size_t)M * D * 2);
    _Float16* h3    = (_Float16*)carve((size_t)M * HID * 2);

    (void)hipFuncSetAttribute((const void*)attn_fused_kernel,
                              hipFuncAttributeMaxDynamicSharedMemorySize, ATTN_SMEM);

    // Weight conversions fp32 -> fp16, transposed to N-major.
    {
        int n;
        n = D * 3 * D; convert_transpose_kernel<<<(n + 255) / 256, 256, 0, stream>>>(qkv_w, qkvwT, D, 3 * D);
        n = D * D;     convert_transpose_kernel<<<(n + 255) / 256, 256, 0, stream>>>(proj_w, projwT, D, D);
        n = D * HID;   convert_transpose_kernel<<<(n + 255) / 256, 256, 0, stream>>>(fc1_w, fc1wT, D, HID);
        n = HID * D;   convert_transpose_kernel<<<(n + 255) / 256, 256, 0, stream>>>(fc2_w, fc2wT, HID, D);
    }

    // h1 = LN1(x)
    layernorm_f16_kernel<<<M, 256, 0, stream>>>(x, ln1_g, ln1_b, h1);

    // qkv = h1 @ qkv_w + qkv_b  -> f16
    gemm_wmma_kernel<<<dim3(MT, 3 * D / 64), 256, 0, stream>>>(
        h1, qkvwT, M, D, 3 * D, qkv_b, nullptr, nullptr, qkv16, 0);

    // wa = softmax_over_heads(q k^T * scale) @ v  -> f16 (b,n,h*64+dh)
    attn_fused_kernel<<<dim3((Nn + 15) / 16, Bb), 128, ATTN_SMEM, stream>>>(qkv16, wa16, Nn);

    // xmid = x + wa @ proj_w + proj_b  -> f32
    gemm_wmma_kernel<<<dim3(MT, D / 64), 256, 0, stream>>>(
        wa16, projwT, M, D, D, proj_b, x, xmid, nullptr, 0);

    // h2 = LN2(xmid)
    layernorm_f16_kernel<<<M, 256, 0, stream>>>(xmid, ln2_g, ln2_b, h2);

    // h3 = gelu(h2 @ fc1_w + fc1_b)  -> f16
    gemm_wmma_kernel<<<dim3(MT, HID / 64), 256, 0, stream>>>(
        h2, fc1wT, M, D, HID, fc1_b, nullptr, nullptr, h3, 1);

    // out = xmid + h3 @ fc2_w + fc2_b  -> f32
    gemm_wmma_kernel<<<dim3(MT, D / 64), 256, 0, stream>>>(
        h3, fc2wT, M, HID, D, fc2_b, xmid, (float*)d_out, nullptr, 0);
}
